// OrienCrossAttnBlock_89369679495531
// MI455X (gfx1250) — matrix operand order repaired and checked by reading, not compiled, not run on hardware
//
#include <hip/hip_runtime.h>

// ---------- types ----------
typedef __bf16 v16bf __attribute__((ext_vector_type(16)));
typedef float  v8f   __attribute__((ext_vector_type(8)));
typedef unsigned short u16;
typedef unsigned int   u32;

union Frag { uint4 u[2]; v16bf v; };

__device__ inline u16 f2bf(float f) {
  union { float f; u32 u; } c; c.f = f;
  u32 u = c.u;
  u32 r = u + 0x7fffu + ((u >> 16) & 1u);   // round-to-nearest-even
  return (u16)(r >> 16);
}
__device__ inline float bf2f(u16 h) {
  union { u32 u; float f; } c; c.u = ((u32)h) << 16;
  return c.f;
}

template <typename T> __device__ inline T to_out(float v);
template <> __device__ inline float to_out<float>(float v) { return v; }
template <> __device__ inline u16   to_out<u16>(float v)   { return f2bf(v); }

// ---------- CDNA5 async global -> LDS copy (ASYNCcnt path, ISA 15.18.3) ----------
// LDS generic pointers carry the LDS byte offset in bits[31:0] (aperture rule),
// so truncation gives the ds address the async unit wants.
__device__ inline void async_cp16(void* lds, const void* gmem) {
  u32 loff = (u32)(size_t)lds;
  asm volatile("global_load_async_to_lds_b128 %0, %1, off"
               :: "v"(loff), "v"(gmem) : "memory");
}
__device__ inline void wait_async0() {
  asm volatile("s_wait_asynccnt 0x0" ::: "memory");
}

// ---------- pad + f32->bf16 convert ----------
__global__ void convert_pad_kernel(const float* __restrict__ in, u16* __restrict__ out,
                                   int R, int Cin, int Cout, int total) {
  int idx = blockIdx.x * 256 + threadIdx.x;
  if (idx >= total) return;
  int r = idx / Cout, c = idx % Cout;
  float v = (r < R && c < Cin) ? in[(size_t)r * Cin + c] : 0.f;
  out[idx] = f2bf(v);
}

// ---------- fused transpose + layernorm of x ----------
// x[B,256,4096] -> xf f32 [B*4096,256], bf16 into concat[:, 0:256] (ld 512)
__global__ __launch_bounds__(256) void ln_x_kernel(const float* __restrict__ x,
                                                   const float* __restrict__ g,
                                                   const float* __restrict__ b,
                                                   float* __restrict__ xf,
                                                   u16* __restrict__ concat) {
  const int HW = 4096, C = 256;
  __shared__ float tile[32][C + 4];
  __shared__ float red1[32][8], red2[32][8];
  __shared__ float smean[32], srstd[32];
  int tid = threadIdx.x;
  int blk = blockIdx.x;
  int bidx = blk >> 7;              // 128 blocks of 32 tokens per batch image
  int hw0 = (blk & 127) << 5;
  for (int i = 0; i < 32; ++i) {
    int c = i * 8 + (tid >> 5);
    int tok = tid & 31;
    tile[tok][c] = x[((size_t)(bidx * C + c)) * HW + hw0 + tok];
  }
  __syncthreads();
  {
    int tok = tid >> 3, sub = tid & 7;
    float s = 0.f, s2 = 0.f;
    for (int j = 0; j < 32; ++j) { float v = tile[tok][sub * 32 + j]; s += v; s2 += v * v; }
    red1[tok][sub] = s; red2[tok][sub] = s2;
  }
  __syncthreads();
  if ((tid & 7) == 0) {
    int tok = tid >> 3;
    float s = 0.f, s2 = 0.f;
    for (int k = 0; k < 8; ++k) { s += red1[tok][k]; s2 += red2[tok][k]; }
    float mean = s * (1.f / 256.f);
    float var  = s2 * (1.f / 256.f) - mean * mean;
    smean[tok] = mean;
    srstd[tok] = rsqrtf(var + 1e-5f);
  }
  __syncthreads();
  float gl = g[tid], bl = b[tid];
  for (int j = 0; j < 32; ++j) {
    float v = tile[j][tid];
    float y = (v - smean[j]) * srstd[j] * gl + bl;
    size_t row = (size_t)bidx * HW + hw0 + j;
    xf[row * 256 + tid]     = y;
    concat[row * 512 + tid] = f2bf(y);
  }
}

// ---------- WMMA bf16 GEMM: C[M,N] = A[M,K] * B[N,K]^T ----------
// 128x64 block tile, double-buffered LDS filled by async global->LDS copies;
// 8 waves each compute a 32x32 block via 2x2 wmma 16x16x32.
// ACT: 0=none, 1=elu+1, 2=relu
template <typename OT, int ACT>
__global__ __launch_bounds__(256) void gemm_bf16_kernel(
    const u16* __restrict__ A, int lda,
    const u16* __restrict__ B, int ldb,
    OT* __restrict__ C, int ldc, int K) {
  __shared__ __align__(16) u16 As[2][128 * 32];
  __shared__ __align__(16) u16 Bs[2][64 * 32];
  int tid  = threadIdx.x;
  int lane = tid & 31, wave = tid >> 5;
  int wm = wave & 3, wn = wave >> 2;
  size_t m0 = (size_t)blockIdx.y * 128;
  int    n0 = blockIdx.x * 64;

  // per-thread 16B copy slots: 2 for the A tile (512 slots), 1 for the B tile (256 slots)
  int a0row = tid >> 2, ach = (tid & 3) * 8;
  int a1row = a0row + 64;
  const u16* gA0 = A + (m0 + a0row) * lda + ach;
  const u16* gA1 = A + (m0 + a1row) * lda + ach;
  const u16* gB  = B + (size_t)(n0 + a0row) * ldb + ach;

  auto issue = [&](int buf, int k0) {
    async_cp16(&As[buf][a0row * 32 + ach], gA0 + k0);
    async_cp16(&As[buf][a1row * 32 + ach], gA1 + k0);
    async_cp16(&Bs[buf][a0row * 32 + ach], gB + k0);
  };

  v8f acc[2][2];
  const v8f vzero = {0.f, 0.f, 0.f, 0.f, 0.f, 0.f, 0.f, 0.f};
#pragma unroll
  for (int i = 0; i < 2; ++i)
#pragma unroll
    for (int j = 0; j < 2; ++j) acc[i][j] = vzero;

  const int nk = K >> 5;
  issue(0, 0);
  wait_async0();
  __syncthreads();

  int m = lane & 15, hi = lane >> 4;
  for (int kt = 0; kt < nk; ++kt) {
    int buf = kt & 1;
    if (kt + 1 < nk) issue(buf ^ 1, (kt + 1) << 5);   // prefetch overlaps compute

    Frag a[2], bfr[2];
#pragma unroll
    for (int i = 0; i < 2; ++i) {            // A frag: K interleave per ISA layout
      int row = wm * 32 + i * 16 + m;
      a[i].u[0] = *(const uint4*)(&As[buf][row * 32 + hi * 8]);       // K 0..7 / 8..15
      a[i].u[1] = *(const uint4*)(&As[buf][row * 32 + 16 + hi * 8]);  // K 16..23 / 24..31
    }
#pragma unroll
    for (int j = 0; j < 2; ++j) {            // B frag: column n = lane%16, K interleave per ISA layout (same as A)
      int col = wn * 32 + j * 16 + m;
      bfr[j].u[0] = *(const uint4*)(&Bs[buf][col * 32 + hi * 8]);
      bfr[j].u[1] = *(const uint4*)(&Bs[buf][col * 32 + 16 + hi * 8]);
    }
#pragma unroll
    for (int i = 0; i < 2; ++i)
#pragma unroll
      for (int j = 0; j < 2; ++j)
        acc[i][j] = __builtin_amdgcn_wmma_f32_16x16x32_bf16(
            false, a[i].v, false, bfr[j].v, (short)0, acc[i][j], false, false);

    wait_async0();     // next tile landed in the other buffer
    __syncthreads();   // all waves' copies visible; all reads of this buffer done
  }

#pragma unroll
  for (int i = 0; i < 2; ++i) {
    size_t mbase = m0 + wm * 32 + i * 16 + hi * 8;
#pragma unroll
    for (int j = 0; j < 2; ++j) {
      int col = n0 + wn * 32 + j * 16 + m;
#pragma unroll
      for (int r = 0; r < 8; ++r) {
        float v = acc[i][j][r];
        if (ACT == 1) v = v > 0.f ? v + 1.f : __expf(v);   // elu(x)+1
        if (ACT == 2) v = v > 0.f ? v : 0.f;               // relu
        C[(mbase + r) * ldc + col] = to_out<OT>(v);
      }
    }
  }
}

// ---------- KV = K^T V and Ksum over sequence (per b,h) ----------
__global__ __launch_bounds__(256) void kv_kernel(const float* __restrict__ Kact,
                                                 const float* __restrict__ V,
                                                 float* __restrict__ KV,
                                                 float* __restrict__ Ksum) {
  __shared__ float Ks[8][32], Vs[8][32];
  int tid = threadIdx.x;
  int bh = blockIdx.x;               // b*8 + h
  int b = bh >> 3, h = bh & 7;
  int dv = tid & 31, dbase = tid >> 5;
  float acc[4] = {0.f, 0.f, 0.f, 0.f};
  float ksum = 0.f;
  for (int chunk = 0; chunk < 128; ++chunk) {
    int seq = chunk * 8 + dbase;     // 0..1023 over (s,t)
    int s = seq >> 4, t = seq & 15;
    size_t off = ((size_t)(b * 64 + s)) * 4096 + t * 256 + h * 32 + dv;
    Ks[dbase][dv] = Kact[off];
    Vs[dbase][dv] = V[off];
    __syncthreads();
#pragma unroll
    for (int ss = 0; ss < 8; ++ss) {
      float vv = Vs[ss][dv];
#pragma unroll
      for (int q = 0; q < 4; ++q) acc[q] += Ks[ss][dbase + 8 * q] * vv;
    }
    if (tid < 32) {
#pragma unroll
      for (int ss = 0; ss < 8; ++ss) ksum += Ks[ss][tid];
    }
    __syncthreads();
  }
  size_t base = (size_t)bh * 1024;
#pragma unroll
  for (int q = 0; q < 4; ++q) KV[base + (size_t)(dbase + 8 * q) * 32 + dv] = acc[q];
  if (tid < 32) Ksum[(size_t)bh * 32 + tid] = ksum;
}

// ---------- attention apply: msg = (Q @ KV) * Z ----------
__global__ __launch_bounds__(256) void attn_kernel(const u16* __restrict__ Q,
                                                   const float* __restrict__ KV,
                                                   const float* __restrict__ Ksum,
                                                   u16* __restrict__ msg) {
  __shared__ float sKV[8 * 32 * 32];
  __shared__ float sKs[256];
  int tid = threadIdx.x;
  size_t tok = blockIdx.x;
  int b = (int)(tok >> 12);
  for (int i = 0; i < 32; ++i)
    sKV[i * 256 + tid] = KV[(size_t)b * 8192 + i * 256 + tid];
  sKs[tid] = Ksum[b * 256 + tid];
  __syncthreads();
  int h = tid >> 5, lane = tid & 31;
  float q = bf2f(Q[tok * 256 + h * 32 + lane]);
  float d = q * sKs[h * 32 + lane];
  for (int off = 16; off; off >>= 1) d += __shfl_xor(d, off);
  float Z = 1.f / (d + 1e-6f);
  float acc = 0.f;
#pragma unroll
  for (int dd = 0; dd < 32; ++dd) {
    float qd = __shfl(q, dd);
    acc += qd * sKV[(h * 32 + dd) * 32 + lane];
  }
  msg[tok * 256 + h * 32 + lane] = f2bf(acc * Z);
}

// ---------- LN(row) with g1,b1 -> bf16 into concat[:, 256:512] ----------
// one wave per row; vectorized b128 loads + packed b128 bf16 store
__global__ __launch_bounds__(256) void ln_row_kernel(const float* __restrict__ in,
                                                     const float* __restrict__ g,
                                                     const float* __restrict__ b,
                                                     u16* __restrict__ concat) {
  int tid = threadIdx.x;
  int wave = tid >> 5, lane = tid & 31;
  size_t row = (size_t)blockIdx.x * 8 + wave;
  const float4* p = (const float4*)(in + row * 256 + lane * 8);
  float4 v0 = p[0], v1 = p[1];
  float s  = v0.x + v0.y + v0.z + v0.w + v1.x + v1.y + v1.z + v1.w;
  float s2 = v0.x * v0.x + v0.y * v0.y + v0.z * v0.z + v0.w * v0.w +
             v1.x * v1.x + v1.y * v1.y + v1.z * v1.z + v1.w * v1.w;
  for (int off = 16; off; off >>= 1) { s += __shfl_xor(s, off); s2 += __shfl_xor(s2, off); }
  float mean = s * (1.f / 256.f);
  float rstd = rsqrtf(s2 * (1.f / 256.f) - mean * mean + 1e-5f);
  const float4* gp = (const float4*)(g + lane * 8);
  const float4* bp = (const float4*)(b + lane * 8);
  float4 g0 = gp[0], g1v = gp[1], b0 = bp[0], b1v = bp[1];
  union { u16 h[8]; uint4 u; } pk;
  pk.h[0] = f2bf((v0.x - mean) * rstd * g0.x + b0.x);
  pk.h[1] = f2bf((v0.y - mean) * rstd * g0.y + b0.y);
  pk.h[2] = f2bf((v0.z - mean) * rstd * g0.z + b0.z);
  pk.h[3] = f2bf((v0.w - mean) * rstd * g0.w + b0.w);
  pk.h[4] = f2bf((v1.x - mean) * rstd * g1v.x + b1v.x);
  pk.h[5] = f2bf((v1.y - mean) * rstd * g1v.y + b1v.y);
  pk.h[6] = f2bf((v1.z - mean) * rstd * g1v.z + b1v.z);
  pk.h[7] = f2bf((v1.w - mean) * rstd * g1v.w + b1v.w);
  *(uint4*)(concat + row * 512 + 256 + lane * 8) = pk.u;
}

// ---------- final: LN(g2,b2) + residual, transpose back to [B,C,H,W] ----------
__global__ __launch_bounds__(256) void final_kernel(const float* __restrict__ h2,
                                                    const float* __restrict__ g,
                                                    const float* __restrict__ b,
                                                    const float* __restrict__ xf,
                                                    float* __restrict__ out) {
  const int HW = 4096;
  __shared__ float tile[32][256 + 4];
  __shared__ float s1[8], s2[8];
  int tid = threadIdx.x;
  int blk = blockIdx.x;
  int bidx = blk >> 7;
  int hw0 = (blk & 127) << 5;
  float gl = g[tid], bl = b[tid];
  for (int j = 0; j < 32; ++j) {
    size_t row = (size_t)bidx * HW + hw0 + j;
    float v = h2[row * 256 + tid];
    float a = v, a2 = v * v;
    for (int off = 16; off; off >>= 1) { a += __shfl_xor(a, off); a2 += __shfl_xor(a2, off); }
    if ((tid & 31) == 0) { s1[tid >> 5] = a; s2[tid >> 5] = a2; }
    __syncthreads();
    float S = 0.f, S2 = 0.f;
    for (int w = 0; w < 8; ++w) { S += s1[w]; S2 += s2[w]; }
    float mean = S * (1.f / 256.f);
    float rstd = rsqrtf(S2 * (1.f / 256.f) - mean * mean + 1e-5f);
    tile[j][tid] = (v - mean) * rstd * gl + bl + xf[row * 256 + tid];
    __syncthreads();
  }
  for (int i = 0; i < 32; ++i) {
    int c = i * 8 + (tid >> 5);
    int tok = tid & 31;
    out[((size_t)(bidx * 256 + c)) * HW + hw0 + tok] = tile[tok][c];
  }
}

// ---------- workspace layout (bytes) ----------
constexpr size_t O_CONCAT = 0;                                   // bf16 [32768][512]
constexpr size_t O_XF     = O_CONCAT + (size_t)32768 * 512 * 2;  // f32  [32768][256]
constexpr size_t O_BIG    = O_XF     + (size_t)32768 * 256 * 4;  // f32  [32768][256] (msg_m then h2)
constexpr size_t O_Q      = O_BIG    + (size_t)32768 * 256 * 4;  // bf16 [32768][256]
constexpr size_t O_MSG    = O_Q      + (size_t)32768 * 256 * 2;  // bf16 [32768][256]
constexpr size_t O_H      = O_Q;                                 // bf16 [32768][512], aliases dead Q+MSG
constexpr size_t O_SRCA   = O_MSG    + (size_t)32768 * 256 * 2;  // bf16 [512][736] (source padded)
constexpr size_t O_SRC    = O_SRCA   + (size_t)512 * 736 * 2;    // bf16 [512][832]
constexpr size_t O_KACT   = O_SRC    + (size_t)512 * 832 * 2;    // f32  [512][4096]
constexpr size_t O_V      = O_KACT   + (size_t)512 * 4096 * 4;   // f32  [512][4096]
constexpr size_t O_KV     = O_V      + (size_t)512 * 4096 * 4;   // f32  [64][32][32]
constexpr size_t O_KSUM   = O_KV     + (size_t)64 * 1024 * 4;    // f32  [64][32]
constexpr size_t O_WKEY   = O_KSUM   + (size_t)64 * 32 * 4;      // bf16 [832][736]
constexpr size_t O_WQ     = O_WKEY   + (size_t)832 * 736 * 2;    // bf16 [256][256]
constexpr size_t O_WK     = O_WQ     + (size_t)256 * 256 * 2;    // bf16 [4096][832]
constexpr size_t O_WV     = O_WK     + (size_t)4096 * 832 * 2;   // bf16 [4096][832]
constexpr size_t O_WMERGE = O_WV     + (size_t)4096 * 832 * 2;   // bf16 [256][256]
constexpr size_t O_WMLP1  = O_WMERGE + (size_t)256 * 256 * 2;    // bf16 [512][512]
constexpr size_t O_WMLP2  = O_WMLP1  + (size_t)512 * 512 * 2;    // bf16 [256][512]

extern "C" void kernel_launch(void* const* d_in, const int* in_sizes, int n_in,
                              void* d_out, int out_size, void* d_ws, size_t ws_size,
                              hipStream_t stream) {
  const float* x       = (const float*)d_in[0];
  const float* source  = (const float*)d_in[1];
  const float* w_key   = (const float*)d_in[2];
  const float* w_q     = (const float*)d_in[3];
  const float* w_k     = (const float*)d_in[4];
  const float* w_v     = (const float*)d_in[5];
  const float* w_merge = (const float*)d_in[6];
  const float* w_mlp1  = (const float*)d_in[7];
  const float* w_mlp2  = (const float*)d_in[8];
  const float* g1      = (const float*)d_in[9];
  const float* b1      = (const float*)d_in[10];
  const float* g2      = (const float*)d_in[11];
  const float* b2      = (const float*)d_in[12];

  char* ws = (char*)d_ws;
  u16*   concat = (u16*)(ws + O_CONCAT);
  float* xf     = (float*)(ws + O_XF);
  float* big    = (float*)(ws + O_BIG);
  u16*   Qb     = (u16*)(ws + O_Q);
  u16*   msg    = (u16*)(ws + O_MSG);
  u16*   hbuf   = (u16*)(ws + O_H);
  u16*   srcA   = (u16*)(ws + O_SRCA);
  u16*   srcB   = (u16*)(ws + O_SRC);
  float* Kact   = (float*)(ws + O_KACT);
  float* Vb     = (float*)(ws + O_V);
  float* KVb    = (float*)(ws + O_KV);
  float* Ksumb  = (float*)(ws + O_KSUM);
  u16*   wkeyb  = (u16*)(ws + O_WKEY);
  u16*   wqb    = (u16*)(ws + O_WQ);
  u16*   wkb    = (u16*)(ws + O_WK);
  u16*   wvb    = (u16*)(ws + O_WV);
  u16*   wmgb   = (u16*)(ws + O_WMERGE);
  u16*   wm1b   = (u16*)(ws + O_WMLP1);
  u16*   wm2b   = (u16*)(ws + O_WMLP2);

  auto cvt = [&](const float* in, u16* out, int R, int Cin, int Rp, int Cp) {
    int total = Rp * Cp;
    convert_pad_kernel<<<(total + 255) / 256, 256, 0, stream>>>(in, out, R, Cin, Cp, total);
  };
  // pad K to mult-of-32, N to mult-of-64 (zero padding is inert in the GEMMs)
  cvt(source,  srcA, 512, 720, 512, 736);
  cvt(w_key,  wkeyb, 786, 720, 832, 736);
  cvt(w_q,      wqb, 256, 256, 256, 256);
  cvt(w_k,      wkb, 4096, 786, 4096, 832);
  cvt(w_v,      wvb, 4096, 786, 4096, 832);
  cvt(w_merge, wmgb, 256, 256, 256, 256);
  cvt(w_mlp1,  wm1b, 512, 512, 512, 512);
  cvt(w_mlp2,  wm2b, 256, 512, 256, 512);

  // norm1(x) -> xf (f32 residual) + concat[:, :256] (bf16)
  ln_x_kernel<<<1024, 256, 0, stream>>>(x, g1, b1, xf, concat);

  // src = source @ w_key^T          [512 x 832], bf16
  gemm_bf16_kernel<u16, 0><<<dim3(13, 4), 256, 0, stream>>>(srcA, 736, wkeyb, 736, srcB, 832, 736);
  // Q = elu(xf @ w_q^T)+1           [32768 x 256], bf16
  gemm_bf16_kernel<u16, 1><<<dim3(4, 256), 256, 0, stream>>>(concat, 512, wqb, 256, Qb, 256, 256);
  // K = elu(src @ w_k^T)+1          [512 x 4096], f32
  gemm_bf16_kernel<float, 1><<<dim3(64, 4), 256, 0, stream>>>(srcB, 832, wkb, 832, Kact, 4096, 832);
  // V = src @ w_v^T                 [512 x 4096], f32
  gemm_bf16_kernel<float, 0><<<dim3(64, 4), 256, 0, stream>>>(srcB, 832, wvb, 832, Vb, 4096, 832);

  kv_kernel<<<64, 256, 0, stream>>>(Kact, Vb, KVb, Ksumb);
  attn_kernel<<<32768, 256, 0, stream>>>(Qb, KVb, Ksumb, msg);

  // merge GEMM                      [32768 x 256], f32
  gemm_bf16_kernel<float, 0><<<dim3(4, 256), 256, 0, stream>>>(msg, 256, wmgb, 256, big, 256, 256);
  // norm1(msg) -> concat[:, 256:512] (bf16)
  ln_row_kernel<<<4096, 256, 0, stream>>>(big, g1, b1, concat);
  // mlp1 + relu                     [32768 x 512], bf16
  gemm_bf16_kernel<u16, 2><<<dim3(8, 256), 256, 0, stream>>>(concat, 512, wm1b, 512, hbuf, 512, 512);
  // mlp2                            [32768 x 256], f32
  gemm_bf16_kernel<float, 0><<<dim3(4, 256), 256, 0, stream>>>(hbuf, 512, wm2b, 512, big, 256, 512);

  // norm2 + residual + transpose back to [B,C,H,W]
  final_kernel<<<1024, 256, 0, stream>>>(big, g2, b2, xf, (float*)d_out);
}
